// HashNeRF3RGB_40312563040834
// MI455X (gfx1250) — matrix operand order repaired and checked by reading, not compiled, not run on hardware
//
#include <hip/hip_runtime.h>

// ---------------------------------------------------------------------------
// Fused NeRF head for MI455X (gfx1250): wave32, V_WMMA_F32_16X16X32_F16,
// TDM (tensor_load_to_lds) for the block-wide weight image.
// ---------------------------------------------------------------------------

typedef _Float16     v16h __attribute__((ext_vector_type(16)));
typedef _Float16     v8h  __attribute__((ext_vector_type(8)));
typedef _Float16     h2   __attribute__((ext_vector_type(2)));
typedef float        v8f  __attribute__((ext_vector_type(8)));
typedef unsigned int v4u  __attribute__((ext_vector_type(4)));
typedef int          v8i  __attribute__((ext_vector_type(8)));
typedef int          v4i  __attribute__((ext_vector_type(4)));

union AFrag { v16h v; v8h h[2]; };

// Weight image offsets (in _Float16 units). Layout: W^T as [N][Kpad] row-major.
#define OFF_SW0 0        // 64 x 32  (K=3  padded to 32)
#define OFF_SW1 2048     // 64 x 64
#define OFF_SW2 6144     // 16 x 64
#define OFF_RW0 7168     // 64 x 32  (K=18 padded to 32)
#define OFF_RW1 9216     // 64 x 64
#define OFF_RW2 13312    // 64 x 64
#define OFF_GW0 17408
#define OFF_GW1 19456
#define OFF_GW2 23552
#define OFF_BW0 27648
#define OFF_BW1 29696
#define OFF_BW2 33792
#define W_TOTAL 37888    // halfs = 75776 bytes = 9472 x 8B
#define W_QWORDS 9472

#define H_STRIDE 72      // 64 + 8 half padding -> staggers LDS banks
#define PER_WAVE 3072    // inbuf 512 + hA 1152 + hB 1152 + geo 256 (halfs)
#define LDS_BYTES ((W_TOTAL + 8 * PER_WAVE) * 2)   // 124928 B
#define OUT_CH 193

// ---------------------------------------------------------------------------
// Weight prep: f32 row-major (K x N) -> f16 transposed padded [N][Kpad].
// ---------------------------------------------------------------------------
__device__ __forceinline__ void pack_wT(const float* __restrict__ src,
                                        _Float16* __restrict__ dst,
                                        int K, int N, int Kpad,
                                        int tid, int nth) {
  int tot = N * Kpad;
  for (int i = tid; i < tot; i += nth) {
    int n = i / Kpad;
    int k = i - n * Kpad;
    dst[i] = (k < K) ? (_Float16)src[k * N + n] : (_Float16)0.0f;
  }
}

__global__ void prep_weights(const float* sw0, const float* sw1, const float* sw2,
                             const float* rw0, const float* rw1, const float* rw2,
                             const float* gw0, const float* gw1, const float* gw2,
                             const float* bw0, const float* bw1, const float* bw2,
                             _Float16* ws) {
  int tid = blockIdx.x * blockDim.x + threadIdx.x;
  int nth = gridDim.x * blockDim.x;
  pack_wT(sw0, ws + OFF_SW0,  3, 64, 32, tid, nth);
  pack_wT(sw1, ws + OFF_SW1, 64, 64, 64, tid, nth);
  pack_wT(sw2, ws + OFF_SW2, 64, 16, 64, tid, nth);
  pack_wT(rw0, ws + OFF_RW0, 18, 64, 32, tid, nth);
  pack_wT(rw1, ws + OFF_RW1, 64, 64, 64, tid, nth);
  pack_wT(rw2, ws + OFF_RW2, 64, 64, 64, tid, nth);
  pack_wT(gw0, ws + OFF_GW0, 18, 64, 32, tid, nth);
  pack_wT(gw1, ws + OFF_GW1, 64, 64, 64, tid, nth);
  pack_wT(gw2, ws + OFF_GW2, 64, 64, 64, tid, nth);
  pack_wT(bw0, ws + OFF_BW0, 18, 64, 32, tid, nth);
  pack_wT(bw1, ws + OFF_BW1, 64, 64, 64, tid, nth);
  pack_wT(bw2, ws + OFF_BW2, 64, 64, 64, tid, nth);
}

// ---------------------------------------------------------------------------
// Fragment loaders (documented CDNA5 16-bit WMMA VGPR layouts).
// ---------------------------------------------------------------------------
__device__ __forceinline__ v16h load_afrag(const _Float16* act, int actStride,
                                           int k0, int mrow, int hid) {
  AFrag f;
  const _Float16* p = act + mrow * actStride + k0 + hid * 8;
  f.h[0] = *(const v8h*)p;
  f.h[1] = *(const v8h*)(p + 16);
  return f.v;
}

__device__ __forceinline__ v16h load_bfrag(const _Float16* wT, int KP,
                                           int n0, int k0, int mrow, int hid) {
  AFrag f;
  const _Float16* q = wT + (n0 + mrow) * KP + k0 + hid * 8;
  f.h[0] = *(const v8h*)q;
  f.h[1] = *(const v8h*)(q + 16);
  return f.v;
}

// ---------------------------------------------------------------------------
// Dense layer: keep all NN/16 accumulators live, WMMAs issue back-to-back
// (no D->A hazard NOPs), then one cvt + packed-f16-relu store epilogue.
// ReLU is done post-conversion as v_pk_max_num_f16 (1 op / 2 elements);
// relu(cvt(x)) == cvt(relu(x)) since f32->f16 rounding is monotonic.
// ---------------------------------------------------------------------------
template <int KP, int NN, bool RELU>
__device__ __forceinline__ void dense_layer(const _Float16* act, int actStride,
                                            const _Float16* wT,
                                            _Float16* dst, int dstStride,
                                            int lane) {
  const int mrow = lane & 15;
  const int hid  = (lane >> 4) & 1;
  v16h a[KP / 32];
#pragma unroll
  for (int kk = 0; kk < KP / 32; ++kk)
    a[kk] = load_afrag(act, actStride, kk * 32, mrow, hid);

  v8f acc[NN / 16];
#pragma unroll
  for (int c = 0; c < NN / 16; ++c) {
    acc[c] = {};
#pragma unroll
    for (int kk = 0; kk < KP / 32; ++kk) {
      v16h b = load_bfrag(wT, KP, c * 16, kk * 32, mrow, hid);
      acc[c] = __builtin_amdgcn_wmma_f32_16x16x32_f16(false, a[kk], false, b,
                                                      (short)0, acc[c], false, false);
    }
  }
#pragma unroll
  for (int c = 0; c < NN / 16; ++c) {
#pragma unroll
    for (int r = 0; r < 8; r += 2) {
      h2 p;
      p.x = (_Float16)acc[c][r];
      p.y = (_Float16)acc[c][r + 1];
      if (RELU) {
        h2 z = {};
        p = __builtin_elementwise_max(p, z);   // v_pk_max_num_f16
      }
      dst[(r     + hid * 8) * dstStride + c * 16 + mrow] = p.x;
      dst[(r + 1 + hid * 8) * dstStride + c * 16 + mrow] = p.y;
    }
  }
}

// Final color layer: KP x 64, no relu, f32 straight to global out.
template <int KP>
__device__ __forceinline__ void dense_layer_out(const _Float16* act, int actStride,
                                                const _Float16* wT,
                                                float* __restrict__ out,
                                                long row0, int colBase, int lane) {
  const int mrow = lane & 15;
  const int hid  = (lane >> 4) & 1;
  v16h a[KP / 32];
#pragma unroll
  for (int kk = 0; kk < KP / 32; ++kk)
    a[kk] = load_afrag(act, actStride, kk * 32, mrow, hid);

  v8f acc[4];
#pragma unroll
  for (int c = 0; c < 4; ++c) {
    acc[c] = {};
#pragma unroll
    for (int kk = 0; kk < KP / 32; ++kk) {
      v16h b = load_bfrag(wT, KP, c * 16, kk * 32, mrow, hid);
      acc[c] = __builtin_amdgcn_wmma_f32_16x16x32_f16(false, a[kk], false, b,
                                                      (short)0, acc[c], false, false);
    }
  }
#pragma unroll
  for (int c = 0; c < 4; ++c)
#pragma unroll
    for (int r = 0; r < 8; ++r)
      out[(row0 + r + hid * 8) * (long)OUT_CH + colBase + c * 16 + mrow] = acc[c][r];
}

// sigma layer: 64 -> 16 (no relu). D col 0 = sigma (f32 to out[..,192]),
// full 16x16 f16 tile to LDS geo buffer (cols 1..15 = geo_feat).
__device__ __forceinline__ void sigma_layer(const _Float16* act,
                                            const _Float16* wT,
                                            _Float16* geo,
                                            float* __restrict__ out,
                                            long row0, int lane) {
  const int mrow = lane & 15;
  const int hid  = (lane >> 4) & 1;
  v8f acc = {};
#pragma unroll
  for (int kk = 0; kk < 2; ++kk) {
    v16h a = load_afrag(act, H_STRIDE, kk * 32, mrow, hid);
    v16h b = load_bfrag(wT, 64, 0, kk * 32, mrow, hid);
    acc = __builtin_amdgcn_wmma_f32_16x16x32_f16(false, a, false, b,
                                                 (short)0, acc, false, false);
  }
#pragma unroll
  for (int r = 0; r < 8; ++r)
    geo[(r + hid * 8) * 16 + mrow] = (_Float16)acc[r];
  if (mrow == 0) {
#pragma unroll
    for (int r = 0; r < 8; ++r)
      out[(row0 + r + hid * 8) * (long)OUT_CH + 192] = acc[r];
  }
}

// ---------------------------------------------------------------------------
// TDM: one tensor_load_to_lds copies the 75776 B weight image into LDS.
// D# built per CDNA5 ISA (1D tile, 8-byte elements).
// This toolchain exposes the 6-arg builtin:
//   (uint32x4 g0, int32x8 g1, int32x4 g2, int32x4 g3, int32x8 g4, i32 cpol)
// ---------------------------------------------------------------------------
__device__ __forceinline__ void tdm_load_weights(const _Float16* wimg,
                                                 _Float16* smem, int tid) {
#if __has_builtin(__builtin_amdgcn_tensor_load_to_lds)
  if (tid < 32) {                 // wave 0 issues the DMA (EXEC ignored by TDM)
    unsigned lds_addr = (unsigned)(unsigned long long)(void*)smem; // aperture low bits
    unsigned long long ga = (unsigned long long)wimg;
    v4u g0;
    g0[0] = 1u;                                   // count=1, user descriptor
    g0[1] = lds_addr;                             // lds_addr [63:32]
    g0[2] = (unsigned)ga;                         // global_addr [95:64]
    g0[3] = ((unsigned)(ga >> 32) & 0x01FFFFFFu)  // global_addr [120:96]
            | (2u << 30);                         // type=2 ("image")
    v8i g1;
    g1[0] = (int)(3u << 16);                      // wg_mask=0, data_size=3 (8B)
    g1[1] = (int)((W_QWORDS & 0xFFFFu) << 16);    // tensor_dim0[15:0] @bits 48..63
    g1[2] = (int)(((W_QWORDS >> 16) & 0xFFFFu)    // tensor_dim0[31:16]
                  | (1u << 16));                  // tensor_dim1 = 1
    g1[3] = (int)((W_QWORDS & 0xFFFFu) << 16);    // tile_dim0 @bits 112..127
    g1[4] = 0;                                    // tile_dim1/2 unused
    g1[5] = (int)W_QWORDS;                        // tensor_dim0_stride[31:0]
    g1[6] = 0;
    g1[7] = 0;
    v4i gz4 = {0, 0, 0, 0};
    v8i gz8 = {0, 0, 0, 0, 0, 0, 0, 0};
    __builtin_amdgcn_tensor_load_to_lds(g0, g1, gz4, gz4, gz8, 0);
    __builtin_amdgcn_s_wait_tensorcnt(0);
  }
#else
  const uint4* src = (const uint4*)wimg;
  uint4* dst = (uint4*)smem;
  for (int i = tid; i < W_TOTAL / 8; i += 256) dst[i] = src[i];
#endif
}

// ---------------------------------------------------------------------------
// Fused kernel: 8 waves/block, 4 tiles of 16 rows per wave -> 512 rows/block.
// ---------------------------------------------------------------------------
__global__ void __launch_bounds__(256)
nerf_fused(const float* __restrict__ x, const _Float16* __restrict__ wimg,
           float* __restrict__ out, int nrows) {
  extern __shared__ __align__(16) _Float16 smem[];
  const int tid  = threadIdx.x;
  const int lane = tid & 31;
  const int wave = tid >> 5;

  tdm_load_weights(wimg, smem, tid);
  __syncthreads();

  _Float16* inbuf = smem + W_TOTAL + wave * PER_WAVE;  // 16 x 32 staging
  _Float16* hA    = inbuf + 512;                        // 16 x H_STRIDE
  _Float16* hB    = hA + 16 * H_STRIDE;                 // 16 x H_STRIDE
  _Float16* geo   = hB + 16 * H_STRIDE;                 // 16 x 16

  const int w0off[3] = {OFF_RW0, OFF_GW0, OFF_BW0};
  const int w1off[3] = {OFF_RW1, OFF_GW1, OFF_BW1};
  const int w2off[3] = {OFF_RW2, OFF_GW2, OFF_BW2};

  for (int t = 0; t < 4; ++t) {
    const long row0  = (long)blockIdx.x * 512 + wave * 64 + t * 16;
    const bool alive = (row0 + 15) < nrows;

    if (alive) {
      // Prefetch next tile's input rows (global_prefetch_b8).
      __builtin_prefetch(x + (row0 + 16) * 6 + lane, 0, 0);
      // Zero the 16x32 A tile with b128 stores, then drop in the 48 pts.
      v8h z = {};
      for (int i = lane; i < 64; i += 32) ((v8h*)inbuf)[i] = z;
      if (lane < 48) {
        int r = lane / 3, c = lane - r * 3;
        inbuf[r * 32 + c] = (_Float16)x[(row0 + r) * 6 + c];
      }
    }
    __syncthreads();

    // sigma MLP: 3(->32) -> 64 -> 64 -> 16
    if (alive) dense_layer<32, 64, true>(inbuf, 32, smem + OFF_SW0, hA, H_STRIDE, lane);
    __syncthreads();
    if (alive) dense_layer<64, 64, true>(hA, H_STRIDE, smem + OFF_SW1, hB, H_STRIDE, lane);
    __syncthreads();
    if (alive) sigma_layer(hB, smem + OFF_SW2, geo, out, row0, lane);
    __syncthreads();

    // Build color input tile: [views(3) | geo_feat(15) | zeros] as 16x32.
    if (alive) {
      v8h z = {};
      for (int i = lane; i < 64; i += 32) ((v8h*)inbuf)[i] = z;
      if (lane < 48) {
        int r = lane / 3, c = lane - r * 3;
        inbuf[r * 32 + c] = (_Float16)x[(row0 + r) * 6 + 3 + c];
      }
      for (int j = lane; j < 240; j += 32) {        // geo_feat: 16 x 15
        int r = j / 15, c = j - r * 15;
        inbuf[r * 32 + 3 + c] = geo[r * 16 + 1 + c]; // geo col0 = sigma
      }
    }
    __syncthreads();

    // r, g, b MLPs: 18(->32) -> 64 -> 64 -> 64
#pragma unroll
    for (int c3 = 0; c3 < 3; ++c3) {
      if (alive) dense_layer<32, 64, true>(inbuf, 32, smem + w0off[c3], hA, H_STRIDE, lane);
      __syncthreads();
      if (alive) dense_layer<64, 64, true>(hA, H_STRIDE, smem + w1off[c3], hB, H_STRIDE, lane);
      __syncthreads();
      if (alive) dense_layer_out<64>(hB, H_STRIDE, smem + w2off[c3], out, row0, c3 * 64, lane);
      __syncthreads();
    }
  }
}

// ---------------------------------------------------------------------------
extern "C" void kernel_launch(void* const* d_in, const int* in_sizes, int n_in,
                              void* d_out, int out_size, void* d_ws, size_t ws_size,
                              hipStream_t stream) {
  const float* x = (const float*)d_in[0];
  const int nrows = in_sizes[0] / 6;
  _Float16* wimg = (_Float16*)d_ws;

  prep_weights<<<64, 256, 0, stream>>>(
      (const float*)d_in[1], (const float*)d_in[2], (const float*)d_in[3],
      (const float*)d_in[4], (const float*)d_in[5], (const float*)d_in[6],
      (const float*)d_in[7], (const float*)d_in[8], (const float*)d_in[9],
      (const float*)d_in[10], (const float*)d_in[11], (const float*)d_in[12],
      wimg);

  const int nblocks = (nrows + 511) / 512;
  nerf_fused<<<nblocks, 256, LDS_BYTES, stream>>>(x, wimg, (float*)d_out, nrows);
}
